// Psi_30932354465852
// MI455X (gfx1250) — compile-verified, matrix operand order checked by reading, not run on hardware
//
#include <hip/hip_runtime.h>
#include <math.h>

// ---------------------------------------------------------------------------
// Problem constants (match reference)
// ---------------------------------------------------------------------------
#define NC 50000
#define NA 50000
#define DD 256
#define HH 4
#define CH 64
#define EDIM 64
#define E_C2A 300000
#define E_A2C 300000
#define E_T   200000
#define LN_EPS 1e-5f

// ---------------------------------------------------------------------------
// Types (CDNA5 / gfx1250, wave32)
// ---------------------------------------------------------------------------
typedef __bf16 bf16;
typedef bf16  v16bf __attribute__((ext_vector_type(16)));
typedef bf16  v8bf  __attribute__((ext_vector_type(8)));
typedef float v8f   __attribute__((ext_vector_type(8)));
typedef float v4f   __attribute__((ext_vector_type(4)));

// ---------------------------------------------------------------------------
// Weight prep: Wt[n*K + k] = bf16(W[k*N + n])   (N-major, K-contiguous)
// ---------------------------------------------------------------------------
__global__ __launch_bounds__(256)
void convert_transpose(const float* __restrict__ W, bf16* __restrict__ Wt, int K, int N)
{
    int t = blockIdx.x * 256 + threadIdx.x;
    if (t >= N * K) return;
    int n = t / K, k = t - n * K;
    Wt[t] = (bf16)W[(size_t)k * N + n];
}

// ---------------------------------------------------------------------------
// Fragment helpers
// ---------------------------------------------------------------------------
__device__ __forceinline__ v16bf load_bfrag(const bf16* p) {
    v8bf lo = *(const v8bf*)p;
    v8bf hi = *(const v8bf*)(p + 16);
    return __builtin_shufflevector(lo, hi, 0,1,2,3,4,5,6,7,8,9,10,11,12,13,14,15);
}

__device__ __forceinline__ v16bf build_afrag(const float* arow, int koff) {
    const v4f* a0 = (const v4f*)(arow + koff);
    v4f f0 = a0[0];
    v4f f1 = a0[1];
    const v4f* a1 = (const v4f*)(arow + koff + 16);
    v4f f2 = a1[0];
    v4f f3 = a1[1];
    v16bf a;
    #pragma unroll
    for (int j = 0; j < 4; ++j) {
        a[j]      = (bf16)f0[j];
        a[4 + j]  = (bf16)f1[j];
        a[8 + j]  = (bf16)f2[j];
        a[12 + j] = (bf16)f3[j];
    }
    return a;
}

// ---------------------------------------------------------------------------
// GEMM: Out[M,N] (= or +=) act(A[M,K] @ W[K,N] + bias)
// A: f32 row-major.  Wt: bf16 pre-transposed [N][K].
// Block: 128 threads = 4 wave32s. Block tile 128x64; wave tile 32x64
// (2 A fragments x 4 B fragments = 8 WMMA per K-step, B reused twice).
// No LDS / no barriers; EXEC all-ones at every WMMA; M-edge via address clamp.
// ---------------------------------------------------------------------------
#define BM 128
#define BN 64

__global__ __launch_bounds__(128)
void gemm_bias_act(const float* __restrict__ A, const bf16* __restrict__ Wt,
                   const float* __restrict__ bias, float* __restrict__ Out,
                   int M, int K, int N, int act, int acc)
{
    const int wave = threadIdx.x >> 5;
    const int lane = threadIdx.x & 31;
    const int half = lane >> 4;      // K-half selector (ISA 16-bit frag layout)
    const int l    = lane & 15;
    const int m0 = blockIdx.y * BM;
    const int n0 = blockIdx.x * BN;

    const int row0 = m0 + wave * 32 + l;
    const int row1 = row0 + 16;
    const int row0c = (row0 < M) ? row0 : (M - 1);
    const int row1c = (row1 < M) ? row1 : (M - 1);
    const float* arow0 = A + (size_t)row0c * K;
    const float* arow1 = A + (size_t)row1c * K;

    v8f c00 = {}, c01 = {}, c02 = {}, c03 = {};
    v8f c10 = {}, c11 = {}, c12 = {}, c13 = {};

    const size_t brow = (size_t)(n0 + l) * K;

    for (int k0 = 0; k0 < K; k0 += 32) {
        const int koff = k0 + half * 8;

        // ---- issue all fragment loads for this K-step up front ----
        v16bf a0 = build_afrag(arow0, koff);
        v16bf a1 = build_afrag(arow1, koff);
        const bf16* bp = Wt + brow + koff;
        v16bf b0 = load_bfrag(bp);
        v16bf b1 = load_bfrag(bp + 16 * K);
        v16bf b2 = load_bfrag(bp + 32 * K);
        v16bf b3 = load_bfrag(bp + 48 * K);
        if (k0 + 32 < K) {
            __builtin_prefetch(arow0 + k0 + 32, 0, 1);
            __builtin_prefetch(arow1 + k0 + 32, 0, 1);
        }

        // ---- 8 WMMAs, each B fragment reused for both row groups ----
        c00 = __builtin_amdgcn_wmma_f32_16x16x32_bf16(false, a0, false, b0, (short)0, c00, false, false);
        c10 = __builtin_amdgcn_wmma_f32_16x16x32_bf16(false, a1, false, b0, (short)0, c10, false, false);
        c01 = __builtin_amdgcn_wmma_f32_16x16x32_bf16(false, a0, false, b1, (short)0, c01, false, false);
        c11 = __builtin_amdgcn_wmma_f32_16x16x32_bf16(false, a1, false, b1, (short)0, c11, false, false);
        c02 = __builtin_amdgcn_wmma_f32_16x16x32_bf16(false, a0, false, b2, (short)0, c02, false, false);
        c12 = __builtin_amdgcn_wmma_f32_16x16x32_bf16(false, a1, false, b2, (short)0, c12, false, false);
        c03 = __builtin_amdgcn_wmma_f32_16x16x32_bf16(false, a0, false, b3, (short)0, c03, false, false);
        c13 = __builtin_amdgcn_wmma_f32_16x16x32_bf16(false, a1, false, b3, (short)0, c13, false, false);
    }

    // ---- store: lane<16 -> C rows 0..7 (VGPR r), lane>=16 -> rows 8..15 ----
    #pragma unroll
    for (int g = 0; g < 2; ++g) {
        const int rbase = m0 + wave * 32 + g * 16 + half * 8;
        #pragma unroll
        for (int nt = 0; nt < 4; ++nt) {
            const v8f& cc = (g == 0)
                ? ((nt == 0) ? c00 : (nt == 1) ? c01 : (nt == 2) ? c02 : c03)
                : ((nt == 0) ? c10 : (nt == 1) ? c11 : (nt == 2) ? c12 : c13);
            int col = n0 + nt * 16 + l;
            if (col >= N) continue;
            float bv = bias ? bias[col] : 0.0f;
            #pragma unroll
            for (int r = 0; r < 8; ++r) {
                int orow = rbase + r;
                if (orow < M) {
                    float v = cc[r] + bv;
                    if (act == 1) v = v / (1.0f + __expf(-v)); // SiLU
                    size_t idx = (size_t)orow * N + col;
                    Out[idx] = acc ? (Out[idx] + v) : v;
                }
            }
        }
    }
}

// ---------------------------------------------------------------------------
// Ordered-uint encoding for float atomic-max
// ---------------------------------------------------------------------------
__device__ __forceinline__ unsigned encf(float x) {
    unsigned u = __float_as_uint(x);
    return (u & 0x80000000u) ? ~u : (u | 0x80000000u);
}
__device__ __forceinline__ float decf(unsigned k) {
    unsigned u = (k & 0x80000000u) ? (k & 0x7FFFFFFFu) : ~k;
    return __uint_as_float(u);
}

// alpha[e,h] = <q[dst], k[src]+eproj[e]> / sqrt(C);  fused segment-max.
__global__ __launch_bounds__(256)
void attn_score(const float* __restrict__ q, const float* __restrict__ k,
                const float* __restrict__ eproj,
                const int* __restrict__ src, const int* __restrict__ dst,
                float* __restrict__ alpha, unsigned* __restrict__ nodemax, int E)
{
    int t = blockIdx.x * blockDim.x + threadIdx.x;
    if (t >= E * HH) return;
    int e = t >> 2, h = t & 3;
    int s = src[e], d = dst[e];
    const v4f* qr = (const v4f*)(q + (size_t)d * DD + h * CH);
    const v4f* kr = (const v4f*)(k + (size_t)s * DD + h * CH);
    const v4f* er = (const v4f*)(eproj + (size_t)e * DD + h * CH);
    v4f acc4 = {};
    #pragma unroll
    for (int j = 0; j < CH / 4; ++j) acc4 += qr[j] * (kr[j] + er[j]);
    float acc = (acc4[0] + acc4[1] + acc4[2] + acc4[3]) * 0.125f; // 1/sqrt(64)
    alpha[t] = acc;
    atomicMax(&nodemax[d * HH + h], encf(acc));
}

// expo = exp(alpha - segmax); fused segment-sum
__global__ __launch_bounds__(256)
void attn_expsum(const float* __restrict__ alpha, const int* __restrict__ dst,
                 const unsigned* __restrict__ nodemax,
                 float* __restrict__ expo, float* __restrict__ nodesum, int E)
{
    int t = blockIdx.x * blockDim.x + threadIdx.x;
    if (t >= E * HH) return;
    int e = t >> 2, h = t & 3;
    int d = dst[e];
    float m  = decf(nodemax[d * HH + h]);
    float ex = __expf(alpha[t] - m);
    expo[t] = ex;
    atomicAdd(&nodesum[d * HH + h], ex);
}

// out[dst] += (v[src] + eproj) * expo/sum.
// 4 edges per 256-thread block; each thread handles 4 channels (b128 loads).
__global__ __launch_bounds__(256)
void attn_aggregate(const float* __restrict__ v, const float* __restrict__ eproj,
                    const float* __restrict__ expo, const float* __restrict__ nodesum,
                    const int* __restrict__ src, const int* __restrict__ dst,
                    float* __restrict__ out, int E)
{
    int e = blockIdx.x * 4 + (threadIdx.x >> 6);
    if (e >= E) return;
    int c4 = (threadIdx.x & 63) * 4;      // channel base 0..252
    int h = c4 >> 6;
    int s = src[e], d = dst[e];
    float coeff = expo[e * HH + h] / (nodesum[d * HH + h] + 1e-16f);
    v4f vv = *(const v4f*)(v + (size_t)s * DD + c4);
    v4f ee = *(const v4f*)(eproj + (size_t)e * DD + c4);
    v4f val = (vv + ee) * coeff;
    float* o = out + (size_t)d * DD + c4;
    atomicAdd(&o[0], val[0]);
    atomicAdd(&o[1], val[1]);
    atomicAdd(&o[2], val[2]);
    atomicAdd(&o[3], val[3]);
}

// LayerNorm + SiLU, one wave32 per 256-elem row, shuffle reductions
__global__ __launch_bounds__(128)
void ln_silu(const float* __restrict__ in, const float* __restrict__ g,
             const float* __restrict__ b, float* __restrict__ out, int Nrow)
{
    int wave = threadIdx.x >> 5, lane = threadIdx.x & 31;
    int row = blockIdx.x * 4 + wave;
    if (row >= Nrow) return;
    const float* x = in + (size_t)row * DD;
    float vals[8], s = 0.0f, s2 = 0.0f;
    #pragma unroll
    for (int j = 0; j < 8; ++j) {
        float t = x[lane + j * 32];
        vals[j] = t; s += t; s2 += t * t;
    }
    #pragma unroll
    for (int off = 16; off > 0; off >>= 1) {
        s  += __shfl_xor(s,  off, 32);
        s2 += __shfl_xor(s2, off, 32);
    }
    float mean = s * (1.0f / 256.0f);
    float var  = s2 * (1.0f / 256.0f) - mean * mean;
    float inv  = rsqrtf(var + LN_EPS);
    #pragma unroll
    for (int j = 0; j < 8; ++j) {
        int c = lane + j * 32;
        float y = (vals[j] - mean) * inv * g[c] + b[c];
        out[(size_t)row * DD + c] = y / (1.0f + __expf(-y));
    }
}

// ---------------------------------------------------------------------------
// Host orchestration
// ---------------------------------------------------------------------------
struct ConvW {
    const float *bk, *bq, *bs, *bv;          // f32 biases
    const bf16 *We, *Wk, *Wq, *Ws, *Wv;      // pre-transposed bf16 weights
};

static void launch_gemm(hipStream_t s, const float* A, const bf16* Wt, const float* bias,
                        float* Out, int M, int K, int N, int act, int acc) {
    dim3 grid((N + BN - 1) / BN, (M + BM - 1) / BM);
    gemm_bias_act<<<grid, 128, 0, s>>>(A, Wt, bias, Out, M, K, N, act, acc);
}

struct Scratch {
    float *q, *k, *v, *eproj, *alpha, *expo, *nodesum;
    unsigned* nodemax;
};

static void run_attention(hipStream_t s, const ConvW& w,
                          const float* x_src, const float* x_dst,
                          const int* src, const int* dst, const float* ea,
                          int n_src, int n_dst, int E, float* out, const Scratch& sc)
{
    launch_gemm(s, x_dst, w.Wq, w.bq, sc.q, n_dst, DD, DD, 0, 0);
    launch_gemm(s, x_src, w.Wk, w.bk, sc.k, n_src, DD, DD, 0, 0);
    launch_gemm(s, x_src, w.Wv, w.bv, sc.v, n_src, DD, DD, 0, 0);
    launch_gemm(s, ea,    w.We, nullptr, sc.eproj, E, EDIM, DD, 0, 0);

    hipMemsetAsync(sc.nodemax, 0, (size_t)n_dst * HH * sizeof(unsigned), s);
    hipMemsetAsync(sc.nodesum, 0, (size_t)n_dst * HH * sizeof(float), s);

    int th = 256;
    int nb = (E * HH + th - 1) / th;
    attn_score<<<nb, th, 0, s>>>(sc.q, sc.k, sc.eproj, src, dst, sc.alpha, sc.nodemax, E);
    attn_expsum<<<nb, th, 0, s>>>(sc.alpha, dst, sc.nodemax, sc.expo, sc.nodesum, E);
    attn_aggregate<<<(E + 3) / 4, 256, 0, s>>>(sc.v, sc.eproj, sc.expo, sc.nodesum,
                                               src, dst, out, E);
}

extern "C" void kernel_launch(void* const* d_in, const int* in_sizes, int n_in,
                              void* d_out, int out_size, void* d_ws, size_t ws_size,
                              hipStream_t stream)
{
    (void)in_sizes; (void)n_in; (void)out_size; (void)ws_size;

    // top-level inputs (setup_inputs dict order)
    const float* x_curr  = (const float*)d_in[0];
    const float* x_act   = (const float*)d_in[1];
    const int*   src_c2a = (const int*)d_in[2];
    const int*   dst_c2a = (const int*)d_in[3];
    const float* ea_c2a  = (const float*)d_in[4];
    const int*   src_a2c = (const int*)d_in[5];
    const int*   dst_a2c = (const int*)d_in[6];
    const float* ea_a2c  = (const float*)d_in[7];
    const int*   src_t   = (const int*)d_in[8];
    const int*   dst_t   = (const int*)d_in[9];
    const float* ea_t    = (const float*)d_in[10];

    // params pytree, JAX flatten order (dict keys sorted):
    // in_act{W,b}=11,12  in_curr{W,b}=13,14
    // layers[l]: a2c@15+27l, c2a@24+27l, temp@33+27l (each: We,Wk,Wq,Ws,Wv,bk,bq,bs,bv)
    // ln_act{b,g}=69,70  ln_curr{b,g}=71,72  out_act{W,b}=73,74
    const float* inA_b = (const float*)d_in[12];
    const float* inC_b = (const float*)d_in[14];
    const float* lnA_b = (const float*)d_in[69];
    const float* lnA_g = (const float*)d_in[70];
    const float* lnC_b = (const float*)d_in[71];
    const float* lnC_g = (const float*)d_in[72];
    const float* out_b = (const float*)d_in[74];

    // --- workspace carve ---
    char* p = (char*)d_ws;
    auto take = [&](size_t bytes) { char* r = p; p += (bytes + 255) & ~(size_t)255; return r; };

    // pre-transposed bf16 weights (L2-resident, ~3.7 MB total)
    auto prep = [&](int idx, int K) -> bf16* {
        size_t elems = (size_t)K * DD;
        bf16* wt = (bf16*)take(elems * sizeof(bf16));
        int nb = (int)((elems + 255) / 256);
        convert_transpose<<<nb, 256, 0, stream>>>((const float*)d_in[idx], wt, K, DD);
        return wt;
    };
    bf16* inC_Wt = prep(13, DD);
    bf16* inA_Wt = prep(11, DD);
    bf16* out_Wt = prep(73, DD);
    ConvW convs[2][3]; // [layer][0=a2c,1=c2a,2=temp]
    for (int l = 0; l < 2; ++l) {
        int bases[3] = {15 + 27 * l, 24 + 27 * l, 33 + 27 * l};
        for (int ci = 0; ci < 3; ++ci) {
            int b = bases[ci];
            ConvW w;
            w.We = prep(b + 0, EDIM);
            w.Wk = prep(b + 1, DD);
            w.Wq = prep(b + 2, DD);
            w.Ws = prep(b + 3, DD);
            w.Wv = prep(b + 4, DD);
            w.bk = (const float*)d_in[b + 5];
            w.bq = (const float*)d_in[b + 6];
            w.bs = (const float*)d_in[b + 7];
            w.bv = (const float*)d_in[b + 8];
            convs[l][ci] = w;
        }
    }

    const size_t NODEF = (size_t)NA * DD * sizeof(float);
    float* xc    = (float*)take(NODEF);
    float* xa    = (float*)take(NODEF);
    float* new_c = (float*)take(NODEF);
    float* new_a = (float*)take(NODEF);
    Scratch sc;
    sc.q     = (float*)take(NODEF);
    sc.k     = (float*)take(NODEF);
    sc.v     = (float*)take(NODEF);
    sc.eproj = (float*)take((size_t)E_C2A * DD * sizeof(float));
    sc.alpha = (float*)take((size_t)E_C2A * HH * sizeof(float));
    sc.expo  = (float*)take((size_t)E_C2A * HH * sizeof(float));
    sc.nodemax = (unsigned*)take((size_t)NA * HH * sizeof(unsigned));
    sc.nodesum = (float*)take((size_t)NA * HH * sizeof(float));

    // --- input projections: x = silu(x @ W + b) ---
    launch_gemm(stream, x_curr, inC_Wt, inC_b, xc, NC, DD, DD, 1, 0);
    launch_gemm(stream, x_act,  inA_Wt, inA_b, xa, NA, DD, DD, 1, 0);

    for (int l = 0; l < 2; ++l) {
        const ConvW& a2c  = convs[l][0];
        const ConvW& c2a  = convs[l][1];
        const ConvW& temp = convs[l][2];

        // new_a = skip_c2a(xa) + skip_temp(xa) + messages
        launch_gemm(stream, xa, c2a.Ws,  c2a.bs,  new_a, NA, DD, DD, 0, 0);
        launch_gemm(stream, xa, temp.Ws, temp.bs, new_a, NA, DD, DD, 0, 1);
        run_attention(stream, c2a,  xc, xa, src_c2a, dst_c2a, ea_c2a, NC, NA, E_C2A, new_a, sc);
        run_attention(stream, temp, xa, xa, src_t,   dst_t,   ea_t,   NA, NA, E_T,   new_a, sc);

        // new_c = skip_a2c(xc) + messages
        launch_gemm(stream, xc, a2c.Ws, a2c.bs, new_c, NC, DD, DD, 0, 0);
        run_attention(stream, a2c, xa, xc, src_a2c, dst_a2c, ea_a2c, NA, NC, E_A2C, new_c, sc);

        // x = silu(layernorm(new))
        ln_silu<<<(NA + 3) / 4, 128, 0, stream>>>(new_a, lnA_g, lnA_b, xa, NA);
        ln_silu<<<(NC + 3) / 4, 128, 0, stream>>>(new_c, lnC_g, lnC_b, xc, NC);
    }

    // --- output projection ---
    launch_gemm(stream, xa, out_Wt, out_b, (float*)d_out, NA, DD, DD, 0, 0);
}
